// AttnAware_11287174054048
// MI455X (gfx1250) — matrix operand order repaired and checked by reading, not compiled, not run on hardware
//
#include <hip/hip_runtime.h>
#include <hip/hip_bf16.h>
#include <math.h>
#include <stdint.h>

typedef __bf16 bf16;
typedef __attribute__((ext_vector_type(16))) __bf16 v16bf;
typedef __attribute__((ext_vector_type(8)))  __bf16 bf16x8;
typedef __attribute__((ext_vector_type(8)))  float  v8f;
typedef __attribute__((ext_vector_type(4)))  unsigned int v4u;
typedef __attribute__((ext_vector_type(8)))  unsigned int v8u;

static constexpr int Bn = 4, C = 256, NH = 2, HD = 128, N = 4096, C2 = 512;
static constexpr float RATIO = 1.0f / (1.0f + 1e-8f);
static constexpr float INV_SQRT2 = 0.70710678118654752f;
static constexpr float ATTN_SCALE = 0.08838834764831845f; // 128^-0.5

__device__ __forceinline__ float gelu_f(float x) {
  return 0.5f * x * (1.0f + erff(x * 0.70710678118654752f));
}

__device__ __forceinline__ v8f vzero8() {
  v8f z;
#pragma unroll
  for (int i = 0; i < 8; ++i) z[i] = 0.f;
  return z;
}

// Async global->LDS copy of 16 bytes per lane (GLOBAL_LOAD_ASYNC_TO_LDS_B128,
// tracked by ASYNCcnt). ldsOff = byte offset inside the kernel's LDS block.
__device__ __forceinline__ void async_copy_b128(unsigned int ldsOff, const void* gptr) {
  unsigned long long ga = (unsigned long long)(uintptr_t)gptr;
  asm volatile("global_load_async_to_lds_b128 %0, %1, off"
               :: "v"(ldsOff), "v"(ga) : "memory");
}
__device__ __forceinline__ void async_wait0() {
  asm volatile("s_wait_asynccnt 0x0" ::: "memory");
}

// TDM: one-wave DMA of a 2D bf16 tile [rows x 128] from a strided tensor into
// LDS, padded 4 DWORDs per 64-DWORD row (=> 136-element LDS row stride).
// D# layout per CDNA5 ISA (group0 128b, group1 256b).
__device__ __forceinline__ void tdm_load_tile_128xR(unsigned int ldsOff,
                                                    const void* gptr,
                                                    unsigned int rows,
                                                    unsigned int tensorRows) {
  unsigned long long ga = (unsigned long long)(uintptr_t)gptr;
  v4u g0;
  g0[0] = 1u;                                            // count=1 user D#
  g0[1] = ldsOff;                                        // lds_addr
  g0[2] = (unsigned int)(ga & 0xffffffffu);              // global_addr lo
  g0[3] = (unsigned int)((ga >> 32) & 0x01ffffffu) | 0x80000000u; // hi | type=2
  v8u g1;
  g1[0] = (1u << 16)      // data_size = 2 bytes
        | (1u << 20)      // pad_enable
        | (5u << 22)      // pad_interval: 64 DWORDs (one 128-elem bf16 row)
        | (3u << 25);     // pad_amount: 4 DWORDs (16B) -> 136-elem stride
  g1[1] = (128u & 0xffffu) << 16;                 // tensor_dim0 lo16 (=128)
  g1[2] = (tensorRows & 0xffffu) << 16;           // dim0 hi | tensor_dim1 lo
  g1[3] = (tensorRows >> 16) | (128u << 16);      // tensor_dim1 hi | tile_dim0
  g1[4] = rows & 0xffffu;                         // tile_dim1 | tile_dim2=0
  g1[5] = 128u;                                   // tensor_dim0_stride lo
  g1[6] = 0u;                                     // stride0 hi | stride1 lo
  g1[7] = 0u;                                     // stride1 hi
  asm volatile("tensor_load_to_lds %0, %1" :: "s"(g0), "s"(g1) : "memory");
}

// A-matrix (16x32 bf16) fragment: lane half selects K sub-block k = 16*half + s (contiguous, same as B).
__device__ __forceinline__ v16bf load_a_frag(const bf16* rowPtr, int half) {
  v16bf a;
#pragma unroll
  for (int v = 0; v < 8; ++v) {
    int kb = half * 16 + 2 * v;
    a[2 * v]     = rowPtr[kb];
    a[2 * v + 1] = rowPtr[kb + 1];
  }
  return a;
}

// B-matrix (32x16 bf16) fragment from a [n][k] (k-contiguous) LDS tile.
__device__ __forceinline__ v16bf load_b_frag(const bf16* colPtr, int half) {
  v16bf b;
#pragma unroll
  for (int v = 0; v < 8; ++v) {
    int kb = half * 16 + 2 * v;
    b[2 * v]     = colPtr[kb];
    b[2 * v + 1] = colPtr[kb + 1];
  }
  return b;
}

// ---------------------------------------------------------------- weight cvt
__global__ void cvt_bf16_kernel(const float* __restrict__ src, bf16* __restrict__ dst, int n) {
  int i = blockIdx.x * blockDim.x + threadIdx.x;
  if (i < n) dst[i] = (bf16)src[i];
}

// -------------------------------------------------- pixnorm+gelu on x (C=256)
__global__ __launch_bounds__(256)
void prep_x_kernel(const float* __restrict__ x, const float* __restrict__ aq,
                   const float* __restrict__ ak, bf16* __restrict__ gq,
                   bf16* __restrict__ gk, bf16* __restrict__ vh,
                   float* __restrict__ cat) {
  __shared__ float ssq[64];
  const int b = blockIdx.y;
  const int n = blockIdx.x * 64 + (threadIdx.x & 63);
  const int t = threadIdx.x;
  const float* xb = x + (size_t)b * C * N;
  if (t < 64) ssq[t] = 0.f;
  __syncthreads();
  float part = 0.f;
  for (int c = t >> 6; c < C; c += 4) { float v = xb[(size_t)c * N + n]; part += v * v; }
  atomicAdd(&ssq[t & 63], part);
  __syncthreads();
  if (t < 64) ssq[t] = rsqrtf(ssq[t] * (1.0f / C) + 1e-8f);
  __syncthreads();
  const float r = ssq[t & 63];
  for (int c = t >> 6; c < C; c += 4) {
    float v = xb[(size_t)c * N + n];
    float tn = v * r;
    size_t gi = (size_t)b * C * N + (size_t)c * N + n;
    gq[gi] = (bf16)gelu_f(aq[c] * tn);
    gk[gi] = (bf16)gelu_f(ak[c] * tn);
    int h = c >> 7, d = c & 127;
    vh[(((size_t)b * NH + h) * N + n) * HD + d] = (bf16)v;
    cat[((size_t)b * C2 + C + c) * N + n] = v;
  }
}

// ----------------------------------------- pixnorm+gelu on cat (512ch) & h1
__global__ __launch_bounds__(256)
void prep_cat_kernel(const float* __restrict__ cat, const float* __restrict__ a1,
                     bf16* __restrict__ g1, bf16* __restrict__ catb) {
  __shared__ float ssq[64];
  const int b = blockIdx.y;
  const int n = blockIdx.x * 64 + (threadIdx.x & 63);
  const int t = threadIdx.x;
  const float* cb = cat + (size_t)b * C2 * N;
  if (t < 64) ssq[t] = 0.f;
  __syncthreads();
  float part = 0.f;
  for (int c = t >> 6; c < C2; c += 4) { float v = cb[(size_t)c * N + n]; part += v * v; }
  atomicAdd(&ssq[t & 63], part);
  __syncthreads();
  if (t < 64) ssq[t] = rsqrtf(ssq[t] * (1.0f / C2) + 1e-8f);
  __syncthreads();
  const float r = ssq[t & 63];
  for (int c = t >> 6; c < C2; c += 4) {
    float v = cb[(size_t)c * N + n];
    size_t gi = (size_t)b * C2 * N + (size_t)c * N + n;
    g1[gi] = (bf16)gelu_f(a1[c] * v * r);
    catb[gi] = (bf16)v;
  }
}

__global__ __launch_bounds__(256)
void prep_h1_kernel(const float* __restrict__ h1, const float* __restrict__ a2,
                    bf16* __restrict__ g2) {
  __shared__ float ssq[64];
  const int b = blockIdx.y;
  const int n = blockIdx.x * 64 + (threadIdx.x & 63);
  const int t = threadIdx.x;
  const float* hb = h1 + (size_t)b * C * N;
  if (t < 64) ssq[t] = 0.f;
  __syncthreads();
  float part = 0.f;
  for (int c = t >> 6; c < C; c += 4) { float v = hb[(size_t)c * N + n]; part += v * v; }
  atomicAdd(&ssq[t & 63], part);
  __syncthreads();
  if (t < 64) ssq[t] = rsqrtf(ssq[t] * (1.0f / C) + 1e-8f);
  __syncthreads();
  const float r = ssq[t & 63];
  for (int c = t >> 6; c < C; c += 4) {
    float v = hb[(size_t)c * N + n];
    g2[(size_t)b * C * N + (size_t)c * N + n] = (bf16)gelu_f(a2[c] * v * r);
  }
}

// --------------------------------------------------------- WMMA 1x1-conv GEMM
template <int KDIM, int EPI>
__global__ __launch_bounds__(256)
void gemm_wmma_kernel(const bf16* __restrict__ A, const bf16* __restrict__ Bm,
                      const float* __restrict__ bias, float* __restrict__ outF,
                      bf16* __restrict__ outH, const float* __restrict__ shortcut,
                      float ratio) {
  // single LDS block: As [64][40] @0 (5120B), Bs [64][40] @5120
  __shared__ __align__(16) unsigned char smem[10240];
  bf16* As = (bf16*)smem;
  bf16* Bs = (bf16*)(smem + 5120);
  constexpr unsigned AS_OFF = 0;
  constexpr int LSTR = 40;

  const int n0 = blockIdx.x * 64;
  const int m0 = blockIdx.y * 64;
  const int b  = blockIdx.z;
  const int t = threadIdx.x;
  const int lane = t & 31, w = t >> 5;
  const int wm = w >> 1, wn = w & 1;
  const int half = lane >> 4, nl = lane & 15;
  const bf16* Bb = Bm + (size_t)b * KDIM * N;

  v8f acc0 = vzero8(), acc1 = vzero8();
  const int am = t >> 2, ak = (t & 3) * 8;
  const int bk = t >> 3, bnn = (t & 7) * 8;

  for (int k0 = 0; k0 < KDIM; k0 += 32) {
    __syncthreads();
    // A tile 64x32: async global->LDS, 16B per thread
    async_copy_b128(AS_OFF + (unsigned)(am * LSTR + ak) * 2,
                    &A[(size_t)(m0 + am) * KDIM + k0 + ak]);
    // B tile 32x64: manual transpose into [n][k]
    bf16x8 bv = *(const bf16x8*)&Bb[(size_t)(k0 + bk) * N + n0 + bnn];
#pragma unroll
    for (int e = 0; e < 8; ++e) Bs[(bnn + e) * LSTR + bk] = bv[e];
    if (k0 + 32 < KDIM)
      __builtin_prefetch(&Bb[(size_t)(k0 + 32 + bk) * N + n0 + bnn], 0, 1);
    async_wait0();
    __syncthreads();
    v16bf af  = load_a_frag(&As[(wm * 16 + nl) * LSTR], half);
    v16bf bf0 = load_b_frag(&Bs[(wn * 32 + nl) * LSTR], half);
    v16bf bf1 = load_b_frag(&Bs[(wn * 32 + 16 + nl) * LSTR], half);
    acc0 = __builtin_amdgcn_wmma_f32_16x16x32_bf16(false, af, false, bf0, (short)0, acc0, false, false);
    acc1 = __builtin_amdgcn_wmma_f32_16x16x32_bf16(false, af, false, bf1, (short)0, acc1, false, false);
  }

#pragma unroll
  for (int r = 0; r < 8; ++r) {
    const int c = m0 + wm * 16 + r + half * 8;
    const float bsv = bias[c];
    const int n_a = n0 + wn * 32 + nl;
    const int n_b = n_a + 16;
    const float v0 = acc0[r], v1 = acc1[r];
    if (EPI == 0) {
      int h = c >> 7, d = c & 127;
      outH[(((size_t)b * NH + h) * N + n_a) * HD + d] = (bf16)(v0 + bsv);
      outH[(((size_t)b * NH + h) * N + n_b) * HD + d] = (bf16)(v1 + bsv);
    } else if (EPI == 1) {
      outF[((size_t)b * C + c) * N + n_a] = v0 * ratio + bsv;
      outF[((size_t)b * C + c) * N + n_b] = v1 * ratio + bsv;
    } else {
      size_t ia = ((size_t)b * C + c) * N + n_a;
      size_t ib = ((size_t)b * C + c) * N + n_b;
      outF[ia] = (v0 * ratio + bsv + shortcut[ia]) * INV_SQRT2;
      outF[ib] = (v1 * ratio + bsv + shortcut[ib]) * INV_SQRT2;
    }
  }
}

// ---------------------------------------------------------- flash attention
__global__ __launch_bounds__(128)
void flash_attn_kernel(const bf16* __restrict__ Qm, const bf16* __restrict__ Km,
                       const bf16* __restrict__ Vm, float* __restrict__ cat) {
  // single LDS block; byte offsets (all strides keep 16B alignment):
  //   qs [64][136] @ 0       (17408 B)
  //   kt [64][136] @ 17408   (17408 B)
  //   vt [128][72] @ 34816   (18432 B)
  //   pt [4][16][72] @ 53248 ( 9216 B)   total 62464 B
  __shared__ __align__(16) unsigned char smem[62464];
  bf16* qs = (bf16*)smem;
  bf16* kt = (bf16*)(smem + 17408);
  bf16* vt = (bf16*)(smem + 34816);
  bf16* pt = (bf16*)(smem + 53248);
  constexpr unsigned QS_OFF = 0, KT_OFF = 17408;
  constexpr int QSTR = 136, VSTR = 72;

  const int i0 = blockIdx.x * 64;
  const int bh = blockIdx.y;
  const int b = bh >> 1, h = bh & 1;
  const int t = threadIdx.x;
  const int lane = t & 31, w = t >> 5;
  const int half = lane >> 4, nl = lane & 15;
  const bf16* Qb = Qm + (size_t)bh * N * HD;
  const bf16* Kb = Km + (size_t)bh * N * HD;
  const bf16* Vb = Vm + (size_t)bh * N * HD;

  // Q tile: per-lane async global->LDS
#pragma unroll
  for (int i = 0; i < 8; ++i) {
    int vi = t + 128 * i;
    int row = vi >> 4, dc = (vi & 15) * 8;
    async_copy_b128(QS_OFF + (unsigned)(row * QSTR + dc) * 2,
                    &Qb[(size_t)(i0 + row) * HD + dc]);
  }
  async_wait0();
  __syncthreads();

  v8f O[8];
#pragma unroll
  for (int dt = 0; dt < 8; ++dt) O[dt] = vzero8();
  float m_i[8], l_i[8];
#pragma unroll
  for (int r = 0; r < 8; ++r) { m_i[r] = -1e30f; l_i[r] = 0.f; }

  for (int j0 = 0; j0 < N; j0 += 64) {
    __syncthreads();
    // K tile [64][128]: Tensor Data Mover DMA, one descriptor from wave 0
    // (TDM ignores EXEC; per-wave issue, so gate on wave id).
    if (w == 0) {
      tdm_load_tile_128xR(KT_OFF, Kb + (size_t)j0 * HD, 64u, (unsigned)N);
      __builtin_amdgcn_s_wait_tensorcnt((short)0);
    }
    // V tile transposed to [d][j] (B-layout for PV) - manual copy
    for (int i = 0; i < 64; ++i) {
      int idx = t + 128 * i;
      int row = idx >> 7, dc = idx & 127;
      vt[dc * VSTR + row] = Vb[(size_t)(j0 + row) * HD + dc];
    }
    __syncthreads();

    v8f sf[4];
#pragma unroll
    for (int jt = 0; jt < 4; ++jt) sf[jt] = vzero8();
#pragma unroll
    for (int ks = 0; ks < 4; ++ks) {
      v16bf aq = load_a_frag(&qs[(w * 16 + nl) * QSTR + ks * 32], half);
#pragma unroll
      for (int jt = 0; jt < 4; ++jt) {
        v16bf bk = load_b_frag(&kt[(jt * 16 + nl) * QSTR + ks * 32], half);
        sf[jt] = __builtin_amdgcn_wmma_f32_16x16x32_bf16(false, aq, false, bk, (short)0, sf[jt], false, false);
      }
    }

    float nm[8], sc[8], rs[8];
#pragma unroll
    for (int r = 0; r < 8; ++r) {
      float tm = -1e30f;
#pragma unroll
      for (int jt = 0; jt < 4; ++jt) tm = fmaxf(tm, sf[jt][r] * ATTN_SCALE);
      for (int off = 1; off < 16; off <<= 1) tm = fmaxf(tm, __shfl_xor(tm, off, 32));
      float m2 = fmaxf(m_i[r], tm);
      nm[r] = m2; sc[r] = __expf(m_i[r] - m2); m_i[r] = m2; rs[r] = 0.f;
    }
#pragma unroll
    for (int jt = 0; jt < 4; ++jt)
#pragma unroll
      for (int r = 0; r < 8; ++r) {
        float p = __expf(sf[jt][r] * ATTN_SCALE - nm[r]);
        rs[r] += p;
        pt[(w * 16 + r + half * 8) * VSTR + jt * 16 + nl] = (bf16)p;
      }
#pragma unroll
    for (int r = 0; r < 8; ++r) {
      float s = rs[r];
      for (int off = 1; off < 16; off <<= 1) s += __shfl_xor(s, off, 32);
      l_i[r] = l_i[r] * sc[r] + s;
    }
#pragma unroll
    for (int dt = 0; dt < 8; ++dt)
#pragma unroll
      for (int r = 0; r < 8; ++r) O[dt][r] *= sc[r];

    v16bf ap0 = load_a_frag(&pt[(w * 16 + nl) * VSTR], half);
    v16bf ap1 = load_a_frag(&pt[(w * 16 + nl) * VSTR + 32], half);
#pragma unroll
    for (int dt = 0; dt < 8; ++dt) {
      v16bf bv0 = load_b_frag(&vt[(dt * 16 + nl) * VSTR], half);
      O[dt] = __builtin_amdgcn_wmma_f32_16x16x32_bf16(false, ap0, false, bv0, (short)0, O[dt], false, false);
      v16bf bv1 = load_b_frag(&vt[(dt * 16 + nl) * VSTR + 32], half);
      O[dt] = __builtin_amdgcn_wmma_f32_16x16x32_bf16(false, ap1, false, bv1, (short)0, O[dt], false, false);
    }
  }

  float inv[8];
#pragma unroll
  for (int r = 0; r < 8; ++r) inv[r] = 1.f / l_i[r];
#pragma unroll
  for (int dt = 0; dt < 8; ++dt)
#pragma unroll
    for (int r = 0; r < 8; ++r) {
      int d = dt * 16 + nl;
      int tok = i0 + w * 16 + r + half * 8;
      cat[((size_t)b * C2 + h * HD + d) * N + tok] = O[dt][r] * inv[r];
    }
}

// ---------------------------------------------------------------------- host
extern "C" void kernel_launch(void* const* d_in, const int* in_sizes, int n_in,
                              void* d_out, int out_size, void* d_ws, size_t ws_size,
                              hipStream_t stream) {
  const float* x  = (const float*)d_in[0];
  const float* aq = (const float*)d_in[1];
  const float* Wq = (const float*)d_in[2];
  const float* bq = (const float*)d_in[3];
  const float* ak = (const float*)d_in[4];
  const float* Wk = (const float*)d_in[5];
  const float* bk = (const float*)d_in[6];
  const float* a1 = (const float*)d_in[7];
  const float* W1 = (const float*)d_in[8];
  const float* b1 = (const float*)d_in[9];
  const float* a2 = (const float*)d_in[10];
  const float* W2 = (const float*)d_in[11];
  const float* b2 = (const float*)d_in[12];
  const float* Ws = (const float*)d_in[13];
  const float* bs = (const float*)d_in[14];
  float* out = (float*)d_out;

  char* p = (char*)d_ws;
  auto take = [&](size_t bytes) { char* q = p; p += (bytes + 255) & ~(size_t)255; return q; };
  bf16* Wq_b = (bf16*)take((size_t)C * C * 2);
  bf16* Wk_b = (bf16*)take((size_t)C * C * 2);
  bf16* W1_b = (bf16*)take((size_t)C * C2 * 2);
  bf16* W2_b = (bf16*)take((size_t)C * C * 2);
  bf16* Ws_b = (bf16*)take((size_t)C * C2 * 2);
  bf16* gq  = (bf16*)take((size_t)Bn * C * N * 2);
  bf16* gk  = (bf16*)take((size_t)Bn * C * N * 2);
  bf16* qh  = (bf16*)take((size_t)Bn * C * N * 2);
  bf16* kh  = (bf16*)take((size_t)Bn * C * N * 2);
  bf16* vh  = (bf16*)take((size_t)Bn * C * N * 2);
  float* cat = (float*)take((size_t)Bn * C2 * N * 4);
  bf16* catb = (bf16*)take((size_t)Bn * C2 * N * 2);
  bf16* g1  = (bf16*)take((size_t)Bn * C2 * N * 2);
  float* hshort = (float*)take((size_t)Bn * C * N * 4);
  float* h1 = (float*)take((size_t)Bn * C * N * 4);
  bf16* g2  = (bf16*)take((size_t)Bn * C * N * 2);

  cvt_bf16_kernel<<<(C * C + 255) / 256, 256, 0, stream>>>(Wq, Wq_b, C * C);
  cvt_bf16_kernel<<<(C * C + 255) / 256, 256, 0, stream>>>(Wk, Wk_b, C * C);
  cvt_bf16_kernel<<<(C * C2 + 255) / 256, 256, 0, stream>>>(W1, W1_b, C * C2);
  cvt_bf16_kernel<<<(C * C + 255) / 256, 256, 0, stream>>>(W2, W2_b, C * C);
  cvt_bf16_kernel<<<(C * C2 + 255) / 256, 256, 0, stream>>>(Ws, Ws_b, C * C2);

  prep_x_kernel<<<dim3(N / 64, Bn), 256, 0, stream>>>(x, aq, ak, gq, gk, vh, cat);

  gemm_wmma_kernel<C, 0><<<dim3(N / 64, C / 64, Bn), 256, 0, stream>>>(
      Wq_b, gq, bq, nullptr, qh, nullptr, 1.0f);
  gemm_wmma_kernel<C, 0><<<dim3(N / 64, C / 64, Bn), 256, 0, stream>>>(
      Wk_b, gk, bk, nullptr, kh, nullptr, 1.0f);

  flash_attn_kernel<<<dim3(N / 64, Bn * NH), 128, 0, stream>>>(qh, kh, vh, cat);

  prep_cat_kernel<<<dim3(N / 64, Bn), 256, 0, stream>>>(cat, a1, g1, catb);

  gemm_wmma_kernel<C2, 1><<<dim3(N / 64, C / 64, Bn), 256, 0, stream>>>(
      Ws_b, catb, bs, hshort, nullptr, nullptr, RATIO);
  gemm_wmma_kernel<C2, 1><<<dim3(N / 64, C / 64, Bn), 256, 0, stream>>>(
      W1_b, g1, b1, h1, nullptr, nullptr, RATIO);

  prep_h1_kernel<<<dim3(N / 64, Bn), 256, 0, stream>>>(h1, a2, g2);

  gemm_wmma_kernel<C, 2><<<dim3(N / 64, C / 64, Bn), 256, 0, stream>>>(
      W2_b, g2, b2, out, nullptr, hshort, RATIO);
}
